// LlamaDecoderLayer_63522566308480
// MI455X (gfx1250) — matrix operand order repaired and checked by reading, not compiled, not run on hardware
//
#include <hip/hip_runtime.h>

typedef _Float16 half_t;
typedef __attribute__((ext_vector_type(16))) _Float16 v16h;
typedef __attribute__((ext_vector_type(8)))  _Float16 v8h;
typedef __attribute__((ext_vector_type(8)))  float    v8f;

// ---------------------------------------------------------------------------
// WMMA helpers (CDNA5 wave32, 16x16x32 f16 -> f32)
// ---------------------------------------------------------------------------
__device__ inline v8f wmma_f16(v16h a, v16h b, v8f c) {
    return __builtin_amdgcn_wmma_f32_16x16x32_f16(
        /*neg_a=*/false, a, /*neg_b=*/false, b,
        /*c_mod=*/(short)0, c, /*reuse_a=*/false, /*reuse_b=*/false);
}

// A-matrix 16x32 fragment from a row-major [rows, ld] f16 matrix.
// Lane layout (ISA 7.12.2): lanes 0-15 hold M=lane, K=0..7 & 16..23;
// lanes 16-31 hold M=lane-16, K=8..15 & 24..31.
__device__ inline v16h load_fragA(const half_t* __restrict__ base, int ld,
                                  int row0, int k0, int lane) {
    int m    = lane & 15;
    int koff = (lane >> 4) << 3;
    const half_t* p = base + (size_t)(row0 + m) * ld + k0 + koff;
    v8h lo = *(const v8h*)(p);
    v8h hi = *(const v8h*)(p + 16);
    v16h r;
#pragma unroll
    for (int i = 0; i < 8; ++i) { r[i] = lo[i]; r[i + 8] = hi[i]; }
    return r;
}

// B-matrix 32x16 fragment where B[k, n] = Wrow[n0+n][k0+k] from a row-major
// [N, ld] matrix (torch-Linear weight, K contiguous per row).
// Lanes 0-15 hold column n = lane, K=0..7 & 16..23; lanes 16-31 hold K=8..15 & 24..31.
__device__ inline v16h load_fragB(const half_t* __restrict__ base, int ld,
                                  int n0, int k0, int lane) {
    int n  = lane & 15;
    int kk = (lane >> 4) << 3;
    const half_t* p = base + (size_t)(n0 + n) * ld + k0 + kk;
    v8h lo = *(const v8h*)(p);
    v8h hi = *(const v8h*)(p + 16);
    v16h r;
#pragma unroll
    for (int i = 0; i < 8; ++i) { r[i] = lo[i]; r[i + 8] = hi[i]; }
    return r;
}

// ---------------------------------------------------------------------------
// Elementwise helpers
// ---------------------------------------------------------------------------
__global__ void __launch_bounds__(256)
cvt_f32_f16_kernel(const float* __restrict__ in, half_t* __restrict__ out, int n) {
    int i = blockIdx.x * 256 + threadIdx.x;
    if (i < n) out[i] = (half_t)in[i];
}

// v[s*D + d] (f16) -> vt[d*S + s] (f16)
__global__ void __launch_bounds__(256)
transpose_kernel(const half_t* __restrict__ v, half_t* __restrict__ vt,
                 int S, int D) {
    int i = blockIdx.x * 256 + threadIdx.x;
    if (i < S * D) {
        int s = i / D, d = i - s * D;
        vt[(size_t)d * S + s] = v[i];
    }
}

// RMSNorm over last dim, f32 in, f16 out.
__global__ void __launch_bounds__(256)
rmsnorm_kernel(const float* __restrict__ x, const float* __restrict__ w,
               half_t* __restrict__ out, int Dm) {
    int row = blockIdx.x;
    const float* xr = x + (size_t)row * Dm;
    float s = 0.f;
    for (int i = threadIdx.x; i < Dm; i += 256) { float t = xr[i]; s += t * t; }
#pragma unroll
    for (int o = 16; o > 0; o >>= 1) s += __shfl_xor(s, o, 32);
    __shared__ float red[8];
    if ((threadIdx.x & 31) == 0) red[threadIdx.x >> 5] = s;
    __syncthreads();
    float tot = 0.f;
#pragma unroll
    for (int i = 0; i < 8; ++i) tot += red[i];
    float scale = rsqrtf(tot / (float)Dm + 1e-5f);
    for (int i = threadIdx.x; i < Dm; i += 256)
        out[(size_t)row * Dm + i] = (half_t)(xr[i] * scale * w[i]);
}

// ---------------------------------------------------------------------------
// Register-blocked WMMA GEMM: C[M,N] = act(A[M,K] @ W[N,K]^T) (+resid).
// 4 waves/block. Each wave owns a 32x64 output block (2 M-tiles x 4 N-tiles,
// 8 f32 accumulator fragments). Per K-step of 32: 2 A-frags + 4 B-frags feed
// 8 WMMAs -> 1.5 b128 loads per WMMA (vs 4 for an unblocked tile), which is
// what keeps the matrix units fed from L2 (all f16 weights fit in the 192MB
// global L2).
// ---------------------------------------------------------------------------
__global__ void __launch_bounds__(128)
gemm_wmma_kernel(const half_t* __restrict__ A, const half_t* __restrict__ W,
                 float* __restrict__ C, const float* __restrict__ resid,
                 half_t* __restrict__ C16, int M, int N, int K, int act) {
    const int lane = threadIdx.x & 31;
    const int wid  = threadIdx.x >> 5;
    const int mRow = blockIdx.x * 32;
    const int nCol = blockIdx.y * 256 + wid * 64;

    v8f acc[2][4];
#pragma unroll
    for (int i = 0; i < 2; ++i)
#pragma unroll
        for (int j = 0; j < 4; ++j) acc[i][j] = (v8f){};

    for (int k0 = 0; k0 < K; k0 += 32) {
        v16h a0 = load_fragA(A, K, mRow,      k0, lane);
        v16h a1 = load_fragA(A, K, mRow + 16, k0, lane);
        v16h b0 = load_fragB(W, K, nCol,      k0, lane);
        v16h b1 = load_fragB(W, K, nCol + 16, k0, lane);
        v16h b2 = load_fragB(W, K, nCol + 32, k0, lane);
        v16h b3 = load_fragB(W, K, nCol + 48, k0, lane);
        acc[0][0] = wmma_f16(a0, b0, acc[0][0]);
        acc[0][1] = wmma_f16(a0, b1, acc[0][1]);
        acc[0][2] = wmma_f16(a0, b2, acc[0][2]);
        acc[0][3] = wmma_f16(a0, b3, acc[0][3]);
        acc[1][0] = wmma_f16(a1, b0, acc[1][0]);
        acc[1][1] = wmma_f16(a1, b1, acc[1][1]);
        acc[1][2] = wmma_f16(a1, b2, acc[1][2]);
        acc[1][3] = wmma_f16(a1, b3, acc[1][3]);
    }

    const int m15 = lane & 15;
    const int hi8 = (lane >> 4) << 3;
#pragma unroll
    for (int i = 0; i < 2; ++i) {
#pragma unroll
        for (int j = 0; j < 4; ++j) {
            const int n = nCol + j * 16 + m15;
#pragma unroll
            for (int vv = 0; vv < 8; ++vv) {
                int r = mRow + i * 16 + hi8 + vv;
                float x = acc[i][j][vv];
                if (act == 1) x = x / (1.f + __expf(-x));   // SiLU
                if (resid) x += resid[(size_t)r * N + n];
                if (C)   C  [(size_t)r * N + n] = x;
                if (C16) C16[(size_t)r * N + n] = (half_t)x;
            }
        }
    }
}

// ---------------------------------------------------------------------------
// Flash-style attention, HD=64. One wave = 16 query rows of one head.
// scores via 2 WMMAs per 16-key half (K=64 split into 2x32), online softmax
// with shfl_xor butterflies, P staged in LDS to reshape C-layout -> A-layout,
// P@V via 4 WMMAs per 32-key block against V^T [D][S].
// ---------------------------------------------------------------------------
__global__ void __launch_bounds__(128)
attn_kernel(const half_t* __restrict__ q, const half_t* __restrict__ k,
            const half_t* __restrict__ vt, half_t* __restrict__ o,
            int S, int Dm) {
    const int lane = threadIdx.x & 31;
    const int wid  = threadIdx.x >> 5;
    const int h    = blockIdx.y;
    const int qRow = (blockIdx.x * 4 + wid) * 16;
    const int hd0  = h * 64;
    const int m15  = lane & 15;
    const int hi8  = (lane >> 4) << 3;
    const int koff = (lane >> 4) << 3;

    __shared__ alignas(32) half_t Pbuf[4][16 * 32];

    v16h a0 = load_fragA(q, Dm, qRow, hd0,      lane);
    v16h a1 = load_fragA(q, Dm, qRow, hd0 + 32, lane);

    v8f acc0 = {}, acc1 = {}, acc2 = {}, acc3 = {};
    float mrow[8], lrow[8];
#pragma unroll
    for (int i = 0; i < 8; ++i) { mrow[i] = -1e30f; lrow[i] = 0.f; }

    const float sm_scale = 0.125f;  // 1/sqrt(64)

    for (int j0 = 0; j0 < S; j0 += 32) {
        v8f s0 = {}, s1 = {};
        v16h b;
        b = load_fragB(k, Dm, j0,      hd0,      lane); s0 = wmma_f16(a0, b, s0);
        b = load_fragB(k, Dm, j0,      hd0 + 32, lane); s0 = wmma_f16(a1, b, s0);
        b = load_fragB(k, Dm, j0 + 16, hd0,      lane); s1 = wmma_f16(a0, b, s1);
        b = load_fragB(k, Dm, j0 + 16, hd0 + 32, lane); s1 = wmma_f16(a1, b, s1);

        float corr[8];
#pragma unroll
        for (int vv = 0; vv < 8; ++vv) {
            float e0 = s0[vv] * sm_scale;
            float e1 = s1[vv] * sm_scale;
            float mx = fmaxf(e0, e1);
#pragma unroll
            for (int off = 8; off > 0; off >>= 1)
                mx = fmaxf(mx, __shfl_xor(mx, off, 32));
            float mnew = fmaxf(mrow[vv], mx);
            float c  = __expf(mrow[vv] - mnew);
            float p0 = __expf(e0 - mnew);
            float p1 = __expf(e1 - mnew);
            float rs = p0 + p1;
#pragma unroll
            for (int off = 8; off > 0; off >>= 1)
                rs += __shfl_xor(rs, off, 32);
            lrow[vv] = lrow[vv] * c + rs;
            mrow[vv] = mnew;
            corr[vv] = c;
            int prow = vv + hi8;
            Pbuf[wid][prow * 32 + m15]      = (half_t)p0;
            Pbuf[wid][prow * 32 + 16 + m15] = (half_t)p1;
        }
#pragma unroll
        for (int vv = 0; vv < 8; ++vv) {
            acc0[vv] *= corr[vv]; acc1[vv] *= corr[vv];
            acc2[vv] *= corr[vv]; acc3[vv] *= corr[vv];
        }
        __syncthreads();

        // P (16x32) reloaded in A-fragment layout from LDS
        v16h pa;
        {
            const half_t* pr = &Pbuf[wid][m15 * 32 + koff];
            v8h lo = *(const v8h*)(pr);
            v8h hh = *(const v8h*)(pr + 16);
#pragma unroll
            for (int i = 0; i < 8; ++i) { pa[i] = lo[i]; pa[i + 8] = hh[i]; }
        }
        v16h bv;
        bv = load_fragB(vt, S, hd0 +  0, j0, lane); acc0 = wmma_f16(pa, bv, acc0);
        bv = load_fragB(vt, S, hd0 + 16, j0, lane); acc1 = wmma_f16(pa, bv, acc1);
        bv = load_fragB(vt, S, hd0 + 32, j0, lane); acc2 = wmma_f16(pa, bv, acc2);
        bv = load_fragB(vt, S, hd0 + 48, j0, lane); acc3 = wmma_f16(pa, bv, acc3);
        __syncthreads();
    }

    const int rbase = qRow + hi8;
#pragma unroll
    for (int vv = 0; vv < 8; ++vv) {
        int r = rbase + vv;
        float inv = 1.f / lrow[vv];
        o[(size_t)r * Dm + hd0      + m15] = (half_t)(acc0[vv] * inv);
        o[(size_t)r * Dm + hd0 + 16 + m15] = (half_t)(acc1[vv] * inv);
        o[(size_t)r * Dm + hd0 + 32 + m15] = (half_t)(acc2[vv] * inv);
        o[(size_t)r * Dm + hd0 + 48 + m15] = (half_t)(acc3[vv] * inv);
    }
}

// ---------------------------------------------------------------------------
// Host-side orchestration
// ---------------------------------------------------------------------------
extern "C" void kernel_launch(void* const* d_in, const int* in_sizes, int n_in,
                              void* d_out, int out_size, void* d_ws, size_t ws_size,
                              hipStream_t stream) {
    (void)in_sizes; (void)n_in; (void)out_size; (void)ws_size;

    const int S = 4096, D = 768, I = 3072, NH = 12;

    const float* hidden = (const float*)d_in[0];
    const float* wq     = (const float*)d_in[1];
    const float* wk     = (const float*)d_in[2];
    const float* wv     = (const float*)d_in[3];
    const float* wo     = (const float*)d_in[4];
    const float* w_up   = (const float*)d_in[5];
    const float* w_down = (const float*)d_in[6];
    const float* ln1_w  = (const float*)d_in[7];
    const float* ln2_w  = (const float*)d_in[8];
    float* out = (float*)d_out;

    char* ws = (char*)d_ws;
    size_t off = 0;
    auto alloc = [&](size_t bytes) -> void* {
        void* p = ws + off;
        off = (off + bytes + 255) & ~(size_t)255;
        return p;
    };

    half_t* wq16 = (half_t*)alloc((size_t)D * D * 2);
    half_t* wk16 = (half_t*)alloc((size_t)D * D * 2);
    half_t* wv16 = (half_t*)alloc((size_t)D * D * 2);
    half_t* wo16 = (half_t*)alloc((size_t)D * D * 2);
    half_t* wu16 = (half_t*)alloc((size_t)I * D * 2);
    half_t* wd16 = (half_t*)alloc((size_t)D * I * 2);
    half_t* xn16 = (half_t*)alloc((size_t)S * D * 2);
    half_t* q16  = (half_t*)alloc((size_t)S * D * 2);
    half_t* k16  = (half_t*)alloc((size_t)S * D * 2);
    half_t* v16  = (half_t*)alloc((size_t)S * D * 2);
    half_t* vt16 = (half_t*)alloc((size_t)S * D * 2);
    half_t* o16  = (half_t*)alloc((size_t)S * D * 2);
    float*  x1   = (float*) alloc((size_t)S * D * 4);
    half_t* hn16 = (half_t*)alloc((size_t)S * D * 2);
    half_t* h116 = (half_t*)alloc((size_t)S * I * 2);

    // 1. weights -> f16
    auto cvt = [&](const float* src, half_t* dst, int n) {
        cvt_f32_f16_kernel<<<(n + 255) / 256, 256, 0, stream>>>(src, dst, n);
    };
    cvt(wq, wq16, D * D);  cvt(wk, wk16, D * D);
    cvt(wv, wv16, D * D);  cvt(wo, wo16, D * D);
    cvt(w_up, wu16, I * D); cvt(w_down, wd16, D * I);

    // 2. RMSNorm 1
    rmsnorm_kernel<<<S, 256, 0, stream>>>(hidden, ln1_w, xn16, D);

    // 3. QKV projections (f16 out only)
    {
        dim3 grid(S / 32, D / 256);
        gemm_wmma_kernel<<<grid, 128, 0, stream>>>(xn16, wq16, nullptr, nullptr, q16, S, D, D, 0);
        gemm_wmma_kernel<<<grid, 128, 0, stream>>>(xn16, wk16, nullptr, nullptr, k16, S, D, D, 0);
        gemm_wmma_kernel<<<grid, 128, 0, stream>>>(xn16, wv16, nullptr, nullptr, v16, S, D, D, 0);
    }

    // 4. V transpose -> [D][S]
    transpose_kernel<<<(S * D + 255) / 256, 256, 0, stream>>>(v16, vt16, S, D);

    // 5. attention
    {
        dim3 grid(S / 64, NH);
        attn_kernel<<<grid, 128, 0, stream>>>(q16, k16, vt16, o16, S, D);
    }

    // 6. O projection + residual -> x1 (f32)
    {
        dim3 grid(S / 32, D / 256);
        gemm_wmma_kernel<<<grid, 128, 0, stream>>>(o16, wo16, x1, hidden, nullptr, S, D, D, 0);
    }

    // 7. RMSNorm 2
    rmsnorm_kernel<<<S, 256, 0, stream>>>(x1, ln2_w, hn16, D);

    // 8. up-proj + SiLU (f16 out)
    {
        dim3 grid(S / 32, I / 256);
        gemm_wmma_kernel<<<grid, 128, 0, stream>>>(hn16, wu16, nullptr, nullptr, h116, S, I, D, 1);
    }

    // 9. down-proj + residual -> out (f32)
    {
        dim3 grid(S / 32, D / 256);
        gemm_wmma_kernel<<<grid, 128, 0, stream>>>(h116, wd16, out, x1, nullptr, S, D, I, 0);
    }
}
